// Bi_Aug_90950227460849
// MI455X (gfx1250) — matrix operand order repaired and checked by reading, not compiled, not run on hardware
//
#include <hip/hip_runtime.h>
#include <math.h>

#define HH 512
#define WW 512
#define HW (HH*WW)
#define CC 64
#define NP 20000

typedef __attribute__((ext_vector_type(16))) __bf16 v16bf;
typedef __attribute__((ext_vector_type(8)))  __bf16 v8bf;
typedef __attribute__((ext_vector_type(8)))  float  v8f;

__device__ const int SHX[9] = { 0,-1, 1, 0,-1, 1, 0,-1, 1};
__device__ const int SHY[9] = { 0, 0, 0, 1, 1, 1,-1,-1,-1};

// ---------------------------------------------------------------- utilities
__global__ void k_zero_f32(float* __restrict__ p, long n) {
  long i = (long)blockIdx.x * blockDim.x + threadIdx.x;
  long stride = (long)gridDim.x * blockDim.x;
  for (long k = i; k < n; k += stride) p[k] = 0.f;
}

__global__ void k_fill_i32(int* __restrict__ p, int v, long n) {
  long i = (long)blockIdx.x * blockDim.x + threadIdx.x;
  long stride = (long)gridDim.x * blockDim.x;
  for (long k = i; k < n; k += stride) p[k] = v;
}

// grids layout: [sample(2)][side(2)][HW+1], side 0 = li coords, side 1 = ra coords
__global__ void k_build_grid(const int* __restrict__ li_coors,
                             const int* __restrict__ ra_coors,
                             int* __restrict__ grids) {
  int t = blockIdx.x * blockDim.x + threadIdx.x;
  if (t >= 2 * 2 * NP) return;
  int n = t % NP;
  int q = t / NP;
  int side = q & 1, s = q >> 1;
  const int* c = (side == 0 ? li_coors : ra_coors) + (long)(s * NP + n) * 2;
  int r = c[0], cc = c[1];
  if (r >= 0 && r < HH && cc >= 0 && cc < WW)
    grids[(long)(s * 2 + side) * (HW + 1) + r * WW + cc] = n;
}

// --------------------------------------------------- B fragment index helper
// B (32x16) fragment layout in this code (contiguous k, same as A): lane = col%16 + 16*(Kmod32 >= 16),
// element i = Kmod32 % 16, fragment id = n0*2 + (K>=32).
// Stored per 64x64 matrix as 8 fragments x 32 lanes x 16 bf16 = 4096 bf16.
__device__ __forceinline__ int bfrag_index(int t /*K*/, int j /*col*/) {
  int n0 = j >> 4, l15 = j & 15;
  int kb = t >> 5, tp = t & 31, hi = tp >> 4, i = tp & 15;
  return ((n0 * 2 + kb) * 32 + l15 + hi * 16) * 16 + i;
}

// ------------------------------------------- fused-weight precompute (tiny)
// Bmats (bf16, fragment order): [d(2)][4][4096]
//   mat0: (in_wq @ wq)^T   mat1: (in_wk @ wk)^T   mat2: (in_wv @ wv)^T   mat3: out_w^T
// posproj (f32): [d(2)][9][64] = pos_emb @ in_wv^T
__global__ void k_precompute(const float* __restrict__ wq1, const float* __restrict__ wk1,
                             const float* __restrict__ wv1, const float* __restrict__ wq2,
                             const float* __restrict__ wk2, const float* __restrict__ wv2,
                             const float* __restrict__ iw1, const float* __restrict__ iw2,
                             const float* __restrict__ ow1, const float* __restrict__ ow2,
                             const float* __restrict__ pos,
                             unsigned short* __restrict__ Bmats_u,
                             float* __restrict__ posproj) {
  __bf16* Bm = (__bf16*)Bmats_u;
  int d = blockIdx.x, tid = threadIdx.x;
  const float* iw = d ? iw2 : iw1;
  const float* ow = d ? ow2 : ow1;
  const float* wm0 = d ? wq2 : wq1;
  const float* wm1 = d ? wk2 : wk1;
  const float* wm2 = d ? wv2 : wv1;
  for (int e = tid; e < 3 * 4096; e += 256) {
    int mat = e >> 12, r = e & 4095;
    int t = r >> 6, j = r & 63;
    const float* w = (mat == 0) ? wm0 : (mat == 1 ? wm1 : wm2);
    float sum = 0.f;
    for (int k = 0; k < 64; k++) sum += iw[(mat * 64 + j) * 64 + k] * w[k * 64 + t];
    Bm[(long)(d * 4 + mat) * 4096 + bfrag_index(t, j)] = (__bf16)sum;
  }
  for (int e = tid; e < 4096; e += 256) {
    int t = e >> 6, j = e & 63;
    Bm[(long)(d * 4 + 3) * 4096 + bfrag_index(t, j)] = (__bf16)ow[j * 64 + t];
  }
  for (int e = tid; e < 9 * 64; e += 256) {
    int p = e >> 6, c = e & 63;
    float sum = 0.f;
    for (int t = 0; t < 64; t++) sum += pos[p * 64 + t] * iw[(128 + c) * 64 + t];
    posproj[(long)d * 9 * 64 + e] = sum;
  }
}

// ------------------------------------------------ WMMA GEMM core (per block)
// Block = 256 threads = 8 waves; each wave owns a 32x64 output tile
// (2 M-tiles x 4 N-tiles = 16 v_wmma_f32_16x16x32_bf16, B reused across M).
// A staged in LDS in fragment-major order:
//   Afrag[((w*2+mt)*2+kb)*32 + lane] : v16bf  (8 waves * 2 mt * 2 kb = 32 frags)

// Stage 256 rows x 64 cols of f32 X into fragment-order bf16 LDS.
// Chunk = 8 consecutive K of one row -> 8 contiguous bf16 of one lane fragment
// -> converted and written with a single 16B LDS store.
__device__ __forceinline__ void stage_A(v16bf* Afrag, const float* __restrict__ X,
                                        int row0, int tid) {
#pragma unroll
  for (int c = 0; c < 8; c++) {
    int ch = tid + 256 * c;           // 2048 chunks = 256 rows * 8 groups
    int m = ch >> 3, g = ch & 7;
    int gm = row0 + m;
    float4 f0 = make_float4(0.f, 0.f, 0.f, 0.f), f1 = f0;
    if (gm < NP) {
      const float4* xp = (const float4*)(X + (long)gm * 64 + g * 8);
      f0 = xp[0]; f1 = xp[1];
      if (gm + 256 < NP) __builtin_prefetch(X + (long)(gm + 256) * 64 + g * 8, 0, 1);
    }
    v8bf p;
    p[0] = (__bf16)f0.x; p[1] = (__bf16)f0.y; p[2] = (__bf16)f0.z; p[3] = (__bf16)f0.w;
    p[4] = (__bf16)f1.x; p[5] = (__bf16)f1.y; p[6] = (__bf16)f1.z; p[7] = (__bf16)f1.w;
    int w = m >> 5, mt = (m >> 4) & 1, lm = m & 15;
    int kb = g >> 2, gg = g & 3, hi2 = gg >> 1, ib = gg & 1;
    int frag = ((w * 2 + mt) * 2 + kb) * 32 + lm + hi2 * 16;
    ((v8bf*)Afrag)[frag * 2 + ib] = p;
  }
}

__device__ __forceinline__ void gemm_core(const v16bf* Afrag, const __bf16* __restrict__ Bf,
                                          int wv, int lane, v8f acc[2][4]) {
  v16bf a[2][2];
#pragma unroll
  for (int mt = 0; mt < 2; mt++)
#pragma unroll
    for (int kb = 0; kb < 2; kb++)
      a[mt][kb] = Afrag[((wv * 2 + mt) * 2 + kb) * 32 + lane];
  const v16bf* Bv = (const v16bf*)Bf;
#pragma unroll
  for (int n0 = 0; n0 < 4; n0++) {
    v16bf b0 = Bv[(n0 * 2 + 0) * 32 + lane];
    v16bf b1 = Bv[(n0 * 2 + 1) * 32 + lane];
#pragma unroll
    for (int mt = 0; mt < 2; mt++) {
      acc[mt][n0] = __builtin_amdgcn_wmma_f32_16x16x32_bf16(false, a[mt][0], false, b0,
                                                            (short)0, acc[mt][n0], false, false);
      acc[mt][n0] = __builtin_amdgcn_wmma_f32_16x16x32_bf16(false, a[mt][1], false, b1,
                                                            (short)0, acc[mt][n0], false, false);
    }
  }
}

// qkv layout: [(d*2+s)*3 + which][N][64]  (which: 0=QP(+bias), 1=KB, 2=VB)
__global__ __launch_bounds__(256) void k_gemm_qkv(
    const float* __restrict__ li_feats, const float* __restrict__ ra_feats,
    const unsigned short* __restrict__ Bmats_u,
    const float* __restrict__ in_b1, const float* __restrict__ in_b2,
    float* __restrict__ qkv) {
  const __bf16* Bmats = (const __bf16*)Bmats_u;
  __shared__ v16bf Afrag[32];   // 8 waves * 2 mt * 2 kb = 32 frags... per-lane
  __shared__ v16bf Apad[1024 - 32];  // total 1024 v16bf = 32KB (frag*32lanes)
  (void)Apad;
  v16bf* Af = &Afrag[0];  // indexed as [fragId*32 + lane]

  int sd = blockIdx.y, d = sd >> 1, s = sd & 1, which = blockIdx.z;
  const float* X = (which == 0) ? ((d == 0) ? li_feats : ra_feats)
                                : ((d == 0) ? ra_feats : li_feats);
  X += (long)s * NP * CC;
  const __bf16* Bf = Bmats + (long)(d * 4 + which) * 4096;
  const float* bias = (which == 0) ? ((d == 0) ? in_b1 : in_b2) : nullptr;
  int row0 = blockIdx.x * 256;
  int tid = threadIdx.x;

  stage_A(Af, X, row0, tid);
  __syncthreads();

  int lane = tid & 31, wv = tid >> 5;
  int l15 = lane & 15;
  int hi = (lane >= 16) ? 1 : 0;
  v8f zero = {};
  v8f acc[2][4];
#pragma unroll
  for (int mt = 0; mt < 2; mt++)
#pragma unroll
    for (int n0 = 0; n0 < 4; n0++) acc[mt][n0] = zero;

  gemm_core(Af, Bf, wv, lane, acc);

  float* Y = qkv + (long)(sd * 3 + which) * NP * 64;
#pragma unroll
  for (int mt = 0; mt < 2; mt++) {
    int rb = row0 + wv * 32 + mt * 16 + (hi ? 8 : 0);
#pragma unroll
    for (int n0 = 0; n0 < 4; n0++) {
      int col = n0 * 16 + l15;
      float badd = bias ? bias[col] : 0.f;
#pragma unroll
      for (int rr = 0; rr < 8; rr++) {
        int gm = rb + rr;
        if (gm < NP) Y[(long)gm * 64 + col] = acc[mt][n0][rr] + badd;
      }
    }
  }
}

// ---------------------------------------------- 9-neighbor gather attention
__global__ __launch_bounds__(256) void k_attn(
    const float* __restrict__ qkv, const int* __restrict__ grids,
    const int* __restrict__ li_coors, const int* __restrict__ ra_coors,
    const float* __restrict__ in_b1, const float* __restrict__ in_b2,
    const float* __restrict__ posproj, float* __restrict__ O) {
  int sd = blockIdx.y, d = sd >> 1, s = sd & 1;
  int n = blockIdx.x * blockDim.x + threadIdx.x;
  if (n >= NP) return;
  const int* qc = ((d == 0) ? li_coors : ra_coors) + (long)(s * NP + n) * 2;
  const int* gdb = grids + (long)(s * 2 + ((d == 0) ? 1 : 0)) * (HW + 1);
  const float* inb = (d == 0) ? in_b1 : in_b2;
  const float4* bk4 = (const float4*)(inb + 64);
  const float4* bv4 = (const float4*)(inb + 128);
  const float* KB = qkv + (long)(sd * 3 + 1) * NP * 64;
  const float* VB = qkv + (long)(sd * 3 + 2) * NP * 64;
  int r0 = qc[0], c0 = qc[1];

  float4 q[16];
  const float4* q4 = (const float4*)(qkv + (long)(sd * 3) * NP * 64 + (long)n * 64);
#pragma unroll
  for (int i = 0; i < 16; i++) q[i] = q4[i];

  float sc[4][9];
  int idx9[9];
  for (int j = 0; j < 9; j++) {
    int rr = r0 + SHX[j], cc = c0 + SHY[j];
    int idx = -1;
    if (rr >= 0 && rr < HH && cc >= 0 && cc < WW) idx = gdb[rr * WW + cc];
    idx9[j] = idx;
    const float4* kr = (idx >= 0) ? (const float4*)(KB + (long)idx * 64) : nullptr;
    float h0 = 0.f, h1 = 0.f, h2 = 0.f, h3 = 0.f;
#pragma unroll
    for (int c4 = 0; c4 < 16; c4++) {
      float4 kv = bk4[c4];
      if (kr) { float4 t = kr[c4]; kv.x += t.x; kv.y += t.y; kv.z += t.z; kv.w += t.w; }
      float dp = q[c4].x * kv.x + q[c4].y * kv.y + q[c4].z * kv.z + q[c4].w * kv.w;
      int h = c4 >> 2;
      if (h == 0) h0 += dp; else if (h == 1) h1 += dp; else if (h == 2) h2 += dp; else h3 += dp;
    }
    sc[0][j] = h0 * 0.25f; sc[1][j] = h1 * 0.25f; sc[2][j] = h2 * 0.25f; sc[3][j] = h3 * 0.25f;
  }
#pragma unroll
  for (int h = 0; h < 4; h++) {
    float m = sc[h][0];
    for (int j = 1; j < 9; j++) m = fmaxf(m, sc[h][j]);
    float sum = 0.f;
    for (int j = 0; j < 9; j++) { float e = __expf(sc[h][j] - m); sc[h][j] = e; sum += e; }
    float inv = 1.f / sum;
    for (int j = 0; j < 9; j++) sc[h][j] *= inv;
  }
  float4 o[16];
#pragma unroll
  for (int i = 0; i < 16; i++) o[i] = make_float4(0.f, 0.f, 0.f, 0.f);
  const float* pp = posproj + (long)d * 9 * 64;
  for (int j = 0; j < 9; j++) {
    int idx = idx9[j];
    const float4* vr = (idx >= 0) ? (const float4*)(VB + (long)idx * 64) : nullptr;
    const float4* pr = (const float4*)(pp + j * 64);
#pragma unroll
    for (int c4 = 0; c4 < 16; c4++) {
      float4 v = bv4[c4];
      if (vr) {
        float4 t = vr[c4]; float4 p = pr[c4];
        v.x += t.x + p.x; v.y += t.y + p.y; v.z += t.z + p.z; v.w += t.w + p.w;
      }
      float w = sc[c4 >> 2][j];
      o[c4].x += w * v.x; o[c4].y += w * v.y; o[c4].z += w * v.z; o[c4].w += w * v.w;
    }
  }
  float4* Ow = (float4*)(O + ((long)sd * NP + n) * 64);
#pragma unroll
  for (int i = 0; i < 16; i++) Ow[i] = o[i];
}

// ---------------------------- output projection (WMMA) + PointPillars scatter
__global__ __launch_bounds__(256) void k_outproj_scatter(
    const float* __restrict__ O, const unsigned short* __restrict__ Bmats_u,
    const int* __restrict__ li_coors, const int* __restrict__ ra_coors,
    const float* __restrict__ ob1, const float* __restrict__ ob2,
    float* __restrict__ out) {
  const __bf16* Bmats = (const __bf16*)Bmats_u;
  __shared__ v16bf Af[1024];     // 32 frags * 32 lanes
  __shared__ int lin_s[256];
  int sd = blockIdx.y, d = sd >> 1, s = sd & 1;
  const float* X = O + (long)sd * NP * 64;
  const __bf16* Bf = Bmats + (long)(d * 4 + 3) * 4096;
  const float* bias = (d == 0) ? ob1 : ob2;
  const int* qc = ((d == 0) ? li_coors : ra_coors) + (long)s * NP * 2;
  int row0 = blockIdx.x * 256, tid = threadIdx.x;

  stage_A(Af, X, row0, tid);
  {
    int gm = row0 + tid;
    lin_s[tid] = (gm < NP) ? (qc[gm * 2] * WW + qc[gm * 2 + 1]) : 0;
  }
  __syncthreads();

  int lane = tid & 31, wv = tid >> 5;
  int l15 = lane & 15;
  int hi = (lane >= 16) ? 1 : 0;
  v8f zero = {};
  v8f acc[2][4];
#pragma unroll
  for (int mt = 0; mt < 2; mt++)
#pragma unroll
    for (int n0 = 0; n0 < 4; n0++) acc[mt][n0] = zero;

  gemm_core(Af, Bf, wv, lane, acc);

  float* base = out + ((long)d * 2 + s) * 64 * (long)HW;
#pragma unroll
  for (int mt = 0; mt < 2; mt++) {
    int lrow = wv * 32 + mt * 16 + (hi ? 8 : 0);
#pragma unroll
    for (int n0 = 0; n0 < 4; n0++) {
      int col = n0 * 16 + l15;
      float badd = bias[col];
#pragma unroll
      for (int rr = 0; rr < 8; rr++) {
        int gm = row0 + lrow + rr;
        if (gm < NP) base[(long)col * HW + lin_s[lrow + rr]] = acc[mt][n0][rr] + badd;
      }
    }
  }
}

// ------------------------------------------------------------------ launch
extern "C" void kernel_launch(void* const* d_in, const int* in_sizes, int n_in,
                              void* d_out, int out_size, void* d_ws, size_t ws_size,
                              hipStream_t stream) {
  const float* li_feats = (const float*)d_in[0];
  const int*   li_coors = (const int*)d_in[1];
  const float* ra_feats = (const float*)d_in[2];
  const int*   ra_coors = (const int*)d_in[3];
  const float* pos      = (const float*)d_in[4];
  const float* wq1 = (const float*)d_in[5];
  const float* wk1 = (const float*)d_in[6];
  const float* wv1 = (const float*)d_in[7];
  const float* wq2 = (const float*)d_in[8];
  const float* wk2 = (const float*)d_in[9];
  const float* wv2 = (const float*)d_in[10];
  const float* a1iw = (const float*)d_in[11];
  const float* a1ib = (const float*)d_in[12];
  const float* a1ow = (const float*)d_in[13];
  const float* a1ob = (const float*)d_in[14];
  const float* a2iw = (const float*)d_in[15];
  const float* a2ib = (const float*)d_in[16];
  const float* a2ow = (const float*)d_in[17];
  const float* a2ob = (const float*)d_in[18];

  char* ws = (char*)d_ws;
  size_t off = 0;
  auto wsalloc = [&](size_t bytes) { size_t o = off; off = (off + bytes + 255) & ~(size_t)255; return o; };
  int* grids            = (int*)(ws + wsalloc((size_t)4 * (HW + 1) * sizeof(int)));
  unsigned short* Bmats = (unsigned short*)(ws + wsalloc((size_t)2 * 4 * 4096 * sizeof(unsigned short)));
  float* posproj        = (float*)(ws + wsalloc((size_t)2 * 9 * 64 * sizeof(float)));
  float* qkv            = (float*)(ws + wsalloc((size_t)12 * NP * 64 * sizeof(float)));
  float* Obuf           = (float*)(ws + wsalloc((size_t)4 * NP * 64 * sizeof(float)));

  // 1) zero canvases
  k_zero_f32<<<4096, 256, 0, stream>>>((float*)d_out, (long)out_size);
  // 2) grids = -1
  k_fill_i32<<<2048, 256, 0, stream>>>(grids, -1, (long)4 * (HW + 1));
  // 3) scatter pillar indices
  k_build_grid<<<(2 * 2 * NP + 255) / 256, 256, 0, stream>>>(li_coors, ra_coors, grids);
  // 4) fused weights (bf16, fragment order) + pos projection
  k_precompute<<<2, 256, 0, stream>>>(wq1, wk1, wv1, wq2, wk2, wv2,
                                      a1iw, a2iw, a1ow, a2ow, pos, Bmats, posproj);
  // 5) QP/KB/VB via WMMA
  dim3 g5((NP + 255) / 256, 4, 3);
  k_gemm_qkv<<<g5, 256, 0, stream>>>(li_feats, ra_feats, Bmats, a1ib, a2ib, qkv);
  // 6) 9-neighbor attention
  dim3 g6((NP + 255) / 256, 4);
  k_attn<<<g6, 256, 0, stream>>>(qkv, grids, li_coors, ra_coors, a1ib, a2ib, posproj, Obuf);
  // 7) output projection (WMMA) + scatter into canvas
  dim3 g7((NP + 255) / 256, 4);
  k_outproj_scatter<<<g7, 256, 0, stream>>>(Obuf, Bmats, li_coors, ra_coors, a1ob, a2ob, (float*)d_out);
}
